// Tokenizer_65687229825854
// MI455X (gfx1250) — matrix operand order repaired and checked by reading, not compiled, not run on hardware
//
#include <hip/hip_runtime.h>
#include <hip/hip_bf16.h>

// ---------------------------------------------------------------------------
// VQ tokenizer: nearest-code search via split-bf16 WMMA GEMM + fused argmin.
//   imgs  [8,3,16,128,128] f32 -> patches X [M=32768, D=192]
//   codes [4096,192] f32
//   d(m,n) = |x|^2 + |c|^2 - 2 x.c ; argmin over n; thr 380 -> idx or -1
// Compute-bound (1840 FLOP/byte) => bf16 WMMA with hi/lo split for ~fp32
// accuracy (3 products: hx*hc + hx*lc + lx*hc).
// ---------------------------------------------------------------------------

typedef __bf16 bf16_t;
typedef __attribute__((ext_vector_type(8)))  bf16_t v8bf;
typedef __attribute__((ext_vector_type(16))) bf16_t v16bf;
typedef __attribute__((ext_vector_type(8)))  float  v8f;

#define BATCH 8
#define CHN   3
#define TT    16
#define HH    128
#define WW    128
#define HP    16
#define WP    16
#define DD    192
#define MM    32768
#define NN    4096
#define THRESH 380.0f

// workspace byte offsets (all 256B aligned)
#define OFF_XHI ((size_t)0)
#define OFF_XLO ((size_t)MM * DD * 2)                       // 12,582,912
#define OFF_X2  ((size_t)2 * MM * DD * 2)                   // 25,165,824
#define OFF_CHI (OFF_X2 + (size_t)MM * 4)                   // 25,296,896
#define OFF_CLO (OFF_CHI + (size_t)NN * DD * 2)
#define OFF_C2  (OFF_CLO + (size_t)NN * DD * 2)             // total ~28.5 MB

static __device__ __forceinline__ v16bf cat8(v8bf a, v8bf b) {
    return __builtin_shufflevector(a, b, 0,1,2,3,4,5,6,7,8,9,10,11,12,13,14,15);
}

// ---- stage 1: codes -> bf16 hi/lo + |c|^2 ---------------------------------
__global__ void prep_codes(const float* __restrict__ codes,
                           bf16_t* __restrict__ chi, bf16_t* __restrict__ clo,
                           float* __restrict__ c2) {
    const int n = blockIdx.x;
    const int t = threadIdx.x;            // 64 threads
    __shared__ float red[64];
    float s = 0.0f;
    for (int i = t; i < DD; i += 64) {
        float x = codes[(size_t)n * DD + i];
        bf16_t h = (bf16_t)x;
        bf16_t l = (bf16_t)(x - (float)h);
        chi[(size_t)n * DD + i] = h;
        clo[(size_t)n * DD + i] = l;
        s += x * x;
    }
    red[t] = s;
    __syncthreads();
    for (int o = 32; o > 0; o >>= 1) {
        if (t < o) red[t] += red[t + o];
        __syncthreads();
    }
    if (t == 0) c2[n] = red[0];
}

// ---- stage 2: patch extraction fused with bf16 split + |x|^2 --------------
__global__ void prep_patches(const float* __restrict__ imgs,
                             bf16_t* __restrict__ xhi, bf16_t* __restrict__ xlo,
                             float* __restrict__ x2) {
    const int m = blockIdx.x;             // patch id: ((b*16+t)*16+hp)*16+wp
    const int d = threadIdx.x;            // 192 threads: d = (py*8+px)*3+c
    const int wp = m & 15, hp = (m >> 4) & 15, tt = (m >> 8) & 15, b = m >> 12;
    const int c  = d % 3;
    const int px = (d / 3) & 7;
    const int py = d / 24;
    const int h = hp * 8 + py, w = wp * 8 + px;
    float x = imgs[((((size_t)b * CHN + c) * TT + tt) * HH + h) * WW + w];
    bf16_t hi = (bf16_t)x;
    bf16_t lo = (bf16_t)(x - (float)hi);
    xhi[(size_t)m * DD + d] = hi;
    xlo[(size_t)m * DD + d] = lo;

    __shared__ float red[192];
    red[d] = x * x;
    __syncthreads();
    if (d < 64) red[d] += red[d + 128];
    __syncthreads();
    for (int o = 64; o > 0; o >>= 1) {
        if (d < o) red[d] += red[d + o];
        __syncthreads();
    }
    if (d == 0) x2[m] = red[0];
}

// ---- stage 3: WMMA GEMM + fused argmin ------------------------------------
// 8 waves/block; wave owns 16 M-rows, A fragments (hi+lo, K=192) register-
// resident; loops all 256 N-tiles of 16 codes. 18 v_wmma per tile.
__global__ __launch_bounds__(256) void vq_argmin(
    const bf16_t* __restrict__ xhi, const bf16_t* __restrict__ xlo,
    const float*  __restrict__ x2,
    const bf16_t* __restrict__ chi, const bf16_t* __restrict__ clo,
    const float*  __restrict__ c2,
    const unsigned char* __restrict__ active,
    int* __restrict__ out) {
    const int lane  = threadIdx.x & 31;
    const int wave  = threadIdx.x >> 5;
    const int mbase = blockIdx.x * 128 + wave * 16;
    const int half  = lane >> 4;          // 0: lanes 0-15, 1: lanes 16-31
    const int l15   = lane & 15;

    // A fragment layout (16-bit A 16x32): lane L holds row M=L&15;
    // lanes 0-15: K = k0+0..7 (v0-3) and k0+16..23 (v4-7); lanes 16-31: +8.
    v16bf Ahi[6], Alo[6];
    {
        const size_t rowoff = (size_t)(mbase + l15) * DD + half * 8;
        const bf16_t* ph = xhi + rowoff;
        const bf16_t* pl = xlo + rowoff;
#pragma unroll
        for (int kc = 0; kc < 6; ++kc) {
            Ahi[kc] = cat8(*(const v8bf*)(ph + kc * 32),
                           *(const v8bf*)(ph + kc * 32 + 16));
            Alo[kc] = cat8(*(const v8bf*)(pl + kc * 32),
                           *(const v8bf*)(pl + kc * 32 + 16));
        }
    }

    // C/D layout: VGPR v, lane -> row = mbase + v + 8*half, col = n0 + l15
    float x2v[8];
#pragma unroll
    for (int v = 0; v < 8; ++v) x2v[v] = x2[mbase + half * 8 + v];

    float best[8];
    int   bidx[8];
#pragma unroll
    for (int v = 0; v < 8; ++v) { best[v] = __builtin_inff(); bidx[v] = 0; }

    for (int n0 = 0; n0 < NN; n0 += 16) {
        const int col = n0 + l15;
        const float c2l = active[col] ? c2[col] : __builtin_inff();
        // B fragment layout (16-bit B 32x16): lane holds col = lane&15;
        // lanes 0-15: K = k0+0..7 and k0+16..23; lanes 16-31: +8.
        const bf16_t* bh = chi + (size_t)col * DD + half * 8;
        const bf16_t* bl = clo + (size_t)col * DD + half * 8;
        // prefetch next tile's lines into cache (global_prefetch_b8)
        __builtin_prefetch(bh + (size_t)16 * DD, 0, 1);
        __builtin_prefetch(bl + (size_t)16 * DD, 0, 1);

        v8f acc0 = {};  // hx*hc chain
        v8f acc1 = {};  // hx*lc + lx*hc chain (independent for ILP)
#pragma unroll
        for (int kc = 0; kc < 6; ++kc) {
            v16bf Bhi = cat8(*(const v8bf*)(bh + kc * 32),
                             *(const v8bf*)(bh + kc * 32 + 16));
            v16bf Blo = cat8(*(const v8bf*)(bl + kc * 32),
                             *(const v8bf*)(bl + kc * 32 + 16));
            acc0 = __builtin_amdgcn_wmma_f32_16x16x32_bf16(
                false, Ahi[kc], false, Bhi, (short)0, acc0, false, false);
            acc1 = __builtin_amdgcn_wmma_f32_16x16x32_bf16(
                false, Ahi[kc], false, Blo, (short)0, acc1, false, false);
            acc1 = __builtin_amdgcn_wmma_f32_16x16x32_bf16(
                false, Alo[kc], false, Bhi, (short)0, acc1, false, false);
        }
#pragma unroll
        for (int v = 0; v < 8; ++v) {
            float dist = x2v[v] + c2l - 2.0f * (acc0[v] + acc1[v]);
            if (dist < best[v]) { best[v] = dist; bidx[v] = col; }
        }
    }

    // min-reduce (value, index) across the 16 lanes sharing each row
    // (xor masks 1,2,4,8 stay within each 16-lane half)
#pragma unroll
    for (int v = 0; v < 8; ++v) {
        float bd = best[v];
        int   bi = bidx[v];
#pragma unroll
        for (int off = 1; off < 16; off <<= 1) {
            float od = __shfl_xor(bd, off, 32);
            int   oi = __shfl_xor(bi, off, 32);
            if (od < bd || (od == bd && oi < bi)) { bd = od; bi = oi; }
        }
        if (l15 == 0) {
            out[mbase + half * 8 + v] = (bd <= THRESH) ? bi : -1;
        }
    }
}

// ---------------------------------------------------------------------------
extern "C" void kernel_launch(void* const* d_in, const int* in_sizes, int n_in,
                              void* d_out, int out_size, void* d_ws, size_t ws_size,
                              hipStream_t stream) {
    const float*         imgs   = (const float*)d_in[0];
    const float*         codes  = (const float*)d_in[1];
    const unsigned char* active = (const unsigned char*)d_in[2];
    // d_in[3] = patch_size (8) — fixed by problem geometry above.
    int* out = (int*)d_out;

    char*   ws  = (char*)d_ws;
    bf16_t* xhi = (bf16_t*)(ws + OFF_XHI);
    bf16_t* xlo = (bf16_t*)(ws + OFF_XLO);
    float*  x2  = (float*)(ws + OFF_X2);
    bf16_t* chi = (bf16_t*)(ws + OFF_CHI);
    bf16_t* clo = (bf16_t*)(ws + OFF_CLO);
    float*  c2  = (float*)(ws + OFF_C2);

    prep_codes<<<NN, 64, 0, stream>>>(codes, chi, clo, c2);
    prep_patches<<<MM, 192, 0, stream>>>(imgs, xhi, xlo, x2);
    vq_argmin<<<MM / 128, 256, 0, stream>>>(xhi, xlo, x2, chi, clo, c2, active, out);
}
